// ALiBiAttention_13718125543670
// MI455X (gfx1250) — matrix operand order repaired and checked by reading, not compiled, not run on hardware
//
#include <hip/hip_runtime.h>
#include <hip/hip_bf16.h>
#include <math.h>

typedef __attribute__((ext_vector_type(16))) _Float16 v16h;
typedef __attribute__((ext_vector_type(8)))  float    v8f;
typedef __attribute__((ext_vector_type(4)))  unsigned int v4u;
typedef __attribute__((ext_vector_type(8)))  int      v8i;
typedef __attribute__((ext_vector_type(4)))  int      v4i;

#define WMMA_F16(a, b, c) \
  __builtin_amdgcn_wmma_f32_16x16x32_f16(false, (a), false, (b), (short)0, (c), false, false)

// Build a v16h fragment from two 16-byte LDS chunks.
static __device__ __forceinline__ v16h load_frag(const _Float16* p0, const _Float16* p1) {
  union { v16h v; float4 f[2]; } u;
  u.f[0] = *(const float4*)p0;
  u.f[1] = *(const float4*)p1;
  return u.v;
}

// Generic->LDS byte address: ISA 10.2 maps flat LDS addresses by truncation to addr[31:0].
static __device__ __forceinline__ unsigned lds_offset(const void* p) {
  return (unsigned)(unsigned long long)(uintptr_t)p;
}

// ---------------------------------------------------------------------------
// Tensor Data Mover: 2-D tile load (global -> LDS), f16 elements.
// D# per CDNA5 ISA sec.8: group0 = {count/flags, lds_addr, global_addr, type},
// group1 = {wg_mask|data_size, tensor dims, tile dims, dim0 stride}.
// tile_w elements contiguous per row, tile_h rows, row stride in elements.
// This toolchain exposes the 6-arg builtin: (v4u, v8i, v4i, v4i, v8i, cpol).
// ---------------------------------------------------------------------------
static __device__ __forceinline__ void tdm_load_2d(
    unsigned lds_byte, const void* gptr,
    unsigned tile_w, unsigned tile_h, unsigned stride_elems)
{
  unsigned long long ga = (unsigned long long)(uintptr_t)gptr;
  v4u g0;
  g0[0] = 1u;                                              // count=1, user mode
  g0[1] = lds_byte;                                        // LDS dest (bytes)
  g0[2] = (unsigned)(ga & 0xffffffffull);                  // global_addr[31:0]
  g0[3] = (unsigned)((ga >> 32) & 0x01ffffffull)           // global_addr[56:32]
        | (2u << 30);                                      // type=2 ("image")
  v8i g1;
  g1[0] = (int)(1u << 16);                                 // data_size=2B, wg_mask=0
  g1[1] = (int)(tile_w << 16);                             // tensor_dim0[15:0]
  g1[2] = (int)((tile_w >> 16) | (tile_h << 16));          // tensor_dim0 hi | tensor_dim1 lo
  g1[3] = (int)((tile_h >> 16) | (tile_w << 16));          // tensor_dim1 hi | tile_dim0
  g1[4] = (int)(tile_h & 0xffffu);                         // tile_dim1 (tile_dim2=0)
  g1[5] = (int)stride_elems;                               // tensor_dim0_stride lo32
  g1[6] = 0;                                               // stride0 hi / stride1 lo
  g1[7] = 0;
  v4i z4 = {0, 0, 0, 0};
  v8i z8 = {0, 0, 0, 0, 0, 0, 0, 0};
  __builtin_amdgcn_tensor_load_to_lds(g0, g1, z4, z4, z8, 0);
}

// ---------------------------------------------------------------------------
// f32 -> f16 conversion (grid covers n/4 quads)
// ---------------------------------------------------------------------------
__global__ void cvt_kernel(const float* __restrict__ s, _Float16* __restrict__ d, int n) {
  int i = (blockIdx.x * blockDim.x + threadIdx.x) * 4;
  if (i < n) {
    float4 v = *(const float4*)(s + i);
    _Float16 h4[4] = { (_Float16)v.x, (_Float16)v.y, (_Float16)v.z, (_Float16)v.w };
    *(float2*)(d + i) = *(const float2*)h4;
  }
}

// ---------------------------------------------------------------------------
// NT GEMM: C[m,n] = sum_k A[m,k] * W[n,k] + bias[n]
// A: MxK f16 row-major, W: NxK f16 row-major.
// Block: 256 threads = 8 waves. Tile: BM=128, BN=64, BK=32.
// Tiles staged by the Tensor Data Mover; waves only do WMMA.
// ---------------------------------------------------------------------------
template <bool OUT_F16>
__global__ __launch_bounds__(256) void gemm_nt_kernel(
    const _Float16* __restrict__ A, const _Float16* __restrict__ W,
    const float* __restrict__ bias, void* __restrict__ out,
    int M, int N, int K)
{
  constexpr int BM = 128, BN = 64, BK = 32;
  __shared__ __align__(16) _Float16 As[BM][BK];
  __shared__ __align__(16) _Float16 Ws[BN][BK];

  const int tid  = threadIdx.x;
  const int lane = tid & 31;
  const int wave = tid >> 5;
  const int lh   = lane >> 4;   // lane half (0/1)
  const int lr   = lane & 15;
  const int m0   = blockIdx.y * BM;
  const int n0   = blockIdx.x * BN;

  v8f acc[4] = {};

  for (int k0 = 0; k0 < K; k0 += BK) {
    __syncthreads();  // previous-iteration LDS reads complete
    if (tid == 0) {   // TDM: stage both tiles (EXEC-independent, issued once)
      tdm_load_2d(lds_offset(As), &A[(size_t)m0 * K + k0], BK, BM, (unsigned)K);
      tdm_load_2d(lds_offset(Ws), &W[(size_t)n0 * K + k0], BK, BN, (unsigned)K);
    }
    if (k0 + BK < K) {  // prefetch next tiles into GL2 -> global_prefetch
      __builtin_prefetch(&A[(size_t)(m0 + (tid & 127)) * K + k0 + BK], 0, 0);
      __builtin_prefetch(&W[(size_t)(n0 + (tid & 63)) * K + k0 + BK], 0, 0);
    }
    __builtin_amdgcn_s_wait_tensorcnt(0);
    __syncthreads();

    // A fragment: lane half 0 -> k 0..7 & 16..23, half 1 -> k 8..15 & 24..31
    v16h af = load_frag(&As[wave * 16 + lr][lh * 8],
                        &As[wave * 16 + lr][16 + lh * 8]);
    #pragma unroll
    for (int ns = 0; ns < 4; ++ns) {
      // B fragment: lane = column n (a W row), half 0 -> k 0..7 & 16..23, half 1 -> k 8..15 & 24..31
      v16h bf = load_frag(&Ws[ns * 16 + lr][lh * 8],
                          &Ws[ns * 16 + lr][16 + lh * 8]);
      acc[ns] = WMMA_F16(af, bf, acc[ns]);
    }
  }

  // C layout: VGPR i -> row i + 8*lh, lane lr -> col
  #pragma unroll
  for (int ns = 0; ns < 4; ++ns) {
    int c = n0 + ns * 16 + lr;
    float bv = bias[c];
    #pragma unroll
    for (int i = 0; i < 8; ++i) {
      int r = m0 + wave * 16 + i + lh * 8;
      float val = acc[ns][i] + bv;
      if constexpr (OUT_F16) ((_Float16*)out)[(size_t)r * N + c] = (_Float16)val;
      else                   ((float*)out)[(size_t)r * N + c] = val;
    }
  }
}

// ---------------------------------------------------------------------------
// Flash attention with ALiBi. Grid: (L/64, B*H). Block: 128 threads = 4 waves.
// Wave w owns query rows [qt*64 + w*16, +16). Key tiles of 32.
// Q and K tiles staged by TDM; V staged by threads (needs transpose anyway),
// overlapping with the TDM transfer.
// ---------------------------------------------------------------------------
#define AL_L 2048
#define AL_D 1024
#define AL_H 16

__global__ __launch_bounds__(128) void attn_kernel(
    const _Float16* __restrict__ Qg, const _Float16* __restrict__ Kg,
    const _Float16* __restrict__ Vg, _Float16* __restrict__ Og)
{
  __shared__ __align__(16) _Float16 Qs[64][64];     // 8 KB
  __shared__ __align__(16) _Float16 Ks[32][64];     // 4 KB
  __shared__ __align__(16) _Float16 Vts[64][32];    // 4 KB, transposed [hd][key]
  __shared__ __align__(16) _Float16 Ps[4][16][32];  // 4 KB, per-wave P staging

  const int bh = blockIdx.y;
  const int b  = bh / AL_H, h = bh % AL_H;
  const int qt = blockIdx.x;

  const int tid  = threadIdx.x;
  const int lane = tid & 31;
  const int wave = tid >> 5;
  const int lh   = lane >> 4;
  const int lr   = lane & 15;

  // slopes = 2 ** linspace(-4, -0.3, 16)
  const float slope = exp2f(-4.0f + (float)h * (3.7f / 15.0f));

  // Stage Q tile (64x64) via TDM
  const size_t qbase = ((size_t)(b * AL_L + qt * 64)) * AL_D + h * 64;
  if (tid == 0) tdm_load_2d(lds_offset(Qs), &Qg[qbase], 64, 64, AL_D);
  __builtin_amdgcn_s_wait_tensorcnt(0);
  __syncthreads();

  // Persistent Q fragments for hd chunks [0,32) and [32,64)
  v16h qf[2];
  #pragma unroll
  for (int cc = 0; cc < 2; ++cc)
    qf[cc] = load_frag(&Qs[wave * 16 + lr][cc * 32 + lh * 8],
                       &Qs[wave * 16 + lr][cc * 32 + 16 + lh * 8]);

  float mrow[8], lrow[8];
  v8f acco[4] = {};
  #pragma unroll
  for (int i = 0; i < 8; ++i) { mrow[i] = -INFINITY; lrow[i] = 0.0f; }

  for (int kt = 0; kt < AL_L / 32; ++kt) {
    __syncthreads();  // previous-iteration reads of Ks/Vts are done
    const size_t kb = ((size_t)(b * AL_L + kt * 32)) * AL_D + h * 64;
    if (tid == 0) tdm_load_2d(lds_offset(Ks), &Kg[kb], 64, 32, AL_D);
    // V tile staged + transposed by threads, overlapping the TDM transfer
    for (int idx = tid; idx < 32 * 8; idx += 128) {
      int r = idx >> 3, c = (idx & 7) * 8;
      float4 vv = *(const float4*)&Vg[kb + (size_t)r * AL_D + c];
      const _Float16* ve = (const _Float16*)&vv;
      #pragma unroll
      for (int j = 0; j < 8; ++j) Vts[c + j][r] = ve[j];
    }
    __builtin_amdgcn_s_wait_tensorcnt(0);
    __syncthreads();

    // S = Q . K^T for two 16-key subtiles (K-dim 64 = 2 chained WMMAs)
    v8f s[2];
    #pragma unroll
    for (int sub = 0; sub < 2; ++sub) {
      v8f sv = {};
      #pragma unroll
      for (int cc = 0; cc < 2; ++cc) {
        v16h bf = load_frag(&Ks[sub * 16 + lr][cc * 32 + lh * 8],
                            &Ks[sub * 16 + lr][cc * 32 + 16 + lh * 8]);
        sv = WMMA_F16(qf[cc], bf, sv);
      }
      s[sub] = sv;
    }

    // scale (1/sqrt(64)) + ALiBi key bias (per-lane: key position)
    #pragma unroll
    for (int sub = 0; sub < 2; ++sub) {
      float kbias = -slope * (float)(kt * 32 + sub * 16 + lr);
      #pragma unroll
      for (int i = 0; i < 8; ++i) s[sub][i] = s[sub][i] * 0.125f + kbias;
    }

    // Online softmax. C-layout: element i of lane -> row (i + 8*lh), lane lr -> key.
    // Butterfly over masks 1..8 stays within each 16-lane half.
    float newm[8], sc[8];
    #pragma unroll
    for (int i = 0; i < 8; ++i) {
      float v = fmaxf(s[0][i], s[1][i]);
      #pragma unroll
      for (int mask = 1; mask < 16; mask <<= 1) v = fmaxf(v, __shfl_xor(v, mask, 32));
      newm[i] = fmaxf(mrow[i], v);
      sc[i]   = expf(mrow[i] - newm[i]);
      mrow[i] = newm[i];
    }
    #pragma unroll
    for (int i = 0; i < 8; ++i) {
      float p0 = expf(s[0][i] - newm[i]);
      float p1 = expf(s[1][i] - newm[i]);
      s[0][i] = p0; s[1][i] = p1;
      float rs = p0 + p1;
      #pragma unroll
      for (int mask = 1; mask < 16; mask <<= 1) rs += __shfl_xor(rs, mask, 32);
      lrow[i] = lrow[i] * sc[i] + rs;
      #pragma unroll
      for (int ns = 0; ns < 4; ++ns) acco[ns][i] *= sc[i];
    }

    // Stage P (C-layout) into LDS, re-read in A-layout (wave-private region)
    #pragma unroll
    for (int sub = 0; sub < 2; ++sub)
      #pragma unroll
      for (int i = 0; i < 8; ++i)
        Ps[wave][i + lh * 8][sub * 16 + lr] = (_Float16)s[sub][i];
    asm volatile("s_wait_dscnt 0" ::: "memory");
    v16h pf = load_frag(&Ps[wave][lr][lh * 8], &Ps[wave][lr][16 + lh * 8]);

    // O += P . V  (4 hd-subtiles of 16)
    #pragma unroll
    for (int ns = 0; ns < 4; ++ns) {
      v16h vf = load_frag(&Vts[ns * 16 + lr][lh * 8],
                          &Vts[ns * 16 + lr][16 + lh * 8]);
      acco[ns] = WMMA_F16(pf, vf, acco[ns]);
    }
  }

  // Normalize and store O
  const size_t ob = ((size_t)(b * AL_L + qt * 64)) * AL_D + h * 64;
  #pragma unroll
  for (int ns = 0; ns < 4; ++ns)
    #pragma unroll
    for (int i = 0; i < 8; ++i) {
      int r = wave * 16 + i + lh * 8;
      Og[ob + (size_t)r * AL_D + ns * 16 + lr] = (_Float16)(acco[ns][i] / lrow[i]);
    }
}

// ---------------------------------------------------------------------------
extern "C" void kernel_launch(void* const* d_in, const int* in_sizes, int n_in,
                              void* d_out, int out_size, void* d_ws, size_t ws_size,
                              hipStream_t stream) {
  (void)in_sizes; (void)n_in; (void)out_size; (void)ws_size;

  const float* x  = (const float*)d_in[0];
  const float* Wq = (const float*)d_in[1];
  const float* bq = (const float*)d_in[2];
  const float* Wk = (const float*)d_in[3];
  const float* bk = (const float*)d_in[4];
  const float* Wv = (const float*)d_in[5];
  const float* bv = (const float*)d_in[6];
  const float* Wo = (const float*)d_in[7];
  const float* bo = (const float*)d_in[8];
  float* out = (float*)d_out;

  const int Bb = 2, L = 2048, D = 1024, Mrows = Bb * L;
  char* ws = (char*)d_ws;
  const size_t MB = 1024 * 1024;
  _Float16* x16  = (_Float16*)(ws + 0);       // 8 MB
  _Float16* Wq16 = (_Float16*)(ws + 8  * MB); // 2 MB each
  _Float16* Wk16 = (_Float16*)(ws + 10 * MB);
  _Float16* Wv16 = (_Float16*)(ws + 12 * MB);
  _Float16* Wo16 = (_Float16*)(ws + 14 * MB);
  _Float16* Q16  = (_Float16*)(ws + 16 * MB); // 8 MB each
  _Float16* K16  = (_Float16*)(ws + 24 * MB);
  _Float16* V16  = (_Float16*)(ws + 32 * MB);
  _Float16* A16  = (_Float16*)(ws + 40 * MB);

  // f32 -> f16 staging
  cvt_kernel<<<(Mrows * D / 4 + 255) / 256, 256, 0, stream>>>(x, x16, Mrows * D);
  cvt_kernel<<<(D * D / 4 + 255) / 256, 256, 0, stream>>>(Wq, Wq16, D * D);
  cvt_kernel<<<(D * D / 4 + 255) / 256, 256, 0, stream>>>(Wk, Wk16, D * D);
  cvt_kernel<<<(D * D / 4 + 255) / 256, 256, 0, stream>>>(Wv, Wv16, D * D);
  cvt_kernel<<<(D * D / 4 + 255) / 256, 256, 0, stream>>>(Wo, Wo16, D * D);

  // Q/K/V projections (NT GEMM + bias, f16 out)
  dim3 gg(D / 64, Mrows / 128);
  gemm_nt_kernel<true><<<gg, 256, 0, stream>>>(x16, Wq16, bq, (void*)Q16, Mrows, D, D);
  gemm_nt_kernel<true><<<gg, 256, 0, stream>>>(x16, Wk16, bk, (void*)K16, Mrows, D, D);
  gemm_nt_kernel<true><<<gg, 256, 0, stream>>>(x16, Wv16, bv, (void*)V16, Mrows, D, D);

  // Flash attention with ALiBi
  dim3 ga(L / 64, Bb * AL_H);
  attn_kernel<<<ga, 128, 0, stream>>>(Q16, K16, V16, A16);

  // Output projection (f32 out + bias)
  gemm_nt_kernel<false><<<gg, 256, 0, stream>>>(A16, Wo16, bo, (void*)out, Mrows, D, D);
}
